// GNN_56092272885967
// MI455X (gfx1250) — compile-verified, matrix operand order checked by reading, not run on hardware
//
#include <hip/hip_runtime.h>
#include <hip/hip_bf16.h>

// ---------------------------------------------------------------------------
// MI455X (gfx1250) implementation of the multi-graph GCN + GRU reference.
// wave32; WMMA 16x16x32 f16 (f32 accumulate) for all dense GEMMs with a
// vectorized b128 -> LDS fast path; L2-resident float atomics for the GCN
// edge scatter; single-block LDS scan for the GRU.
// ---------------------------------------------------------------------------

typedef __attribute__((ext_vector_type(16))) _Float16 v16h;
typedef __attribute__((ext_vector_type(4)))  _Float16 h4;
typedef __attribute__((ext_vector_type(8)))  float    v8f;

static __device__ __forceinline__ float sigmoidf_(float x) {
    return 1.f / (1.f + expf(-x));
}

static __device__ __forceinline__ void st4h(_Float16* p, float4 v) {
    h4 t;
    t.x = (_Float16)v.x; t.y = (_Float16)v.y;
    t.z = (_Float16)v.z; t.w = (_Float16)v.w;
    *reinterpret_cast<h4*>(p) = t;   // 64-bit ds_store
}

// ======================= WMMA GEMM =========================================
// C[Mr x Nout] = act( A[Mr x K] * W[Nout x K]^T + bias ), row-major lda/ldc.
// Block: 128 threads = 4 wave32. Block tile 64x64; each wave owns a 16-row
// strip and issues 4 WMMAs per 32-deep K step (A fragment reused 4x).
#define BM 64
#define BN 64
#define BK 32
#define LS 40   // LDS row stride in f16 (80B: 8B-aligned, bank-staggered)

__global__ __launch_bounds__(128) void k_gemm(
    const float* __restrict__ A, int lda,
    const float* __restrict__ W,            // [Nout x K], contiguous
    const float* __restrict__ bias,         // may be nullptr
    float* __restrict__ C, int ldc,
    int Mr, int K, int Nout, int act)
{
    __shared__ _Float16 As[BM][LS];
    __shared__ _Float16 Bs[BN][LS];

    const int tid  = threadIdx.x;
    const int lane = tid & 31;
    const int wv   = tid >> 5;              // 0..3
    const int m0   = blockIdx.x * BM;
    const int n0   = blockIdx.y * BN;

    const bool fullM = (m0 + BM) <= Mr;
    const bool fullN = (n0 + BN) <= Nout;

    v8f acc[4] = {{0.f,0.f,0.f,0.f,0.f,0.f,0.f,0.f},
                  {0.f,0.f,0.f,0.f,0.f,0.f,0.f,0.f},
                  {0.f,0.f,0.f,0.f,0.f,0.f,0.f,0.f},
                  {0.f,0.f,0.f,0.f,0.f,0.f,0.f,0.f}};

    // fast-path staging map: thread t -> row t>>1, 16-col half (t&1)*16
    const int sr = tid >> 1;
    const int sc = (tid & 1) << 4;

    for (int k0 = 0; k0 < K; k0 += BK) {
        const bool fullK = (k0 + BK) <= K;
        if (k0 + BK < K) {     // speculative prefetch of next tiles
            __builtin_prefetch(A + (size_t)(m0 + sr) * lda + (k0 + BK) + sc, 0, 1);
            __builtin_prefetch(W + (size_t)(n0 + sr) * K   + (k0 + BK) + sc, 0, 1);
        }

        // ---- stage A (64x32) ----
        if (fullM && fullK) {
            const float4* ap =
                reinterpret_cast<const float4*>(A + (size_t)(m0 + sr) * lda + k0 + sc);
            float4 a0 = ap[0], a1 = ap[1], a2 = ap[2], a3 = ap[3];
            st4h(&As[sr][sc +  0], a0);
            st4h(&As[sr][sc +  4], a1);
            st4h(&As[sr][sc +  8], a2);
            st4h(&As[sr][sc + 12], a3);
        } else {
            for (int idx = tid; idx < BM * BK; idx += 128) {
                int r = idx >> 5, c = idx & 31;
                int gm = m0 + r, gk = k0 + c;
                float v = (gm < Mr && gk < K) ? A[(size_t)gm * lda + gk] : 0.f;
                As[r][c] = (_Float16)v;
            }
        }
        // ---- stage B (64 rows of W x 32) ----
        if (fullN && fullK) {
            const float4* bp =
                reinterpret_cast<const float4*>(W + (size_t)(n0 + sr) * K + k0 + sc);
            float4 b0 = bp[0], b1 = bp[1], b2 = bp[2], b3 = bp[3];
            st4h(&Bs[sr][sc +  0], b0);
            st4h(&Bs[sr][sc +  4], b1);
            st4h(&Bs[sr][sc +  8], b2);
            st4h(&Bs[sr][sc + 12], b3);
        } else {
            for (int idx = tid; idx < BN * BK; idx += 128) {
                int r = idx >> 5, c = idx & 31;
                int gn = n0 + r, gk = k0 + c;
                float v = (gn < Nout && gk < K) ? W[(size_t)gn * K + gk] : 0.f;
                Bs[r][c] = (_Float16)v;
            }
        }
        __syncthreads();

        // ---- fragments per ISA 7.12.2 (16-bit A 16x32; B analogous) ----
        const int rsel = lane & 15;
        const int kh   = (lane >> 4) * 8;
        const int arow = (wv << 4) + rsel;
        v16h af, bf0, bf1, bf2, bf3;
#pragma unroll
        for (int v = 0; v < 8; ++v) {
            int k = ((v & 4) ? (16 + 2 * (v & 3)) : (2 * v)) + kh;
            af [2*v] = As[arow][k];        af [2*v+1] = As[arow][k+1];
            bf0[2*v] = Bs[rsel      ][k];  bf0[2*v+1] = Bs[rsel      ][k+1];
            bf1[2*v] = Bs[rsel + 16 ][k];  bf1[2*v+1] = Bs[rsel + 16 ][k+1];
            bf2[2*v] = Bs[rsel + 32 ][k];  bf2[2*v+1] = Bs[rsel + 32 ][k+1];
            bf3[2*v] = Bs[rsel + 48 ][k];  bf3[2*v+1] = Bs[rsel + 48 ][k+1];
        }
        acc[0] = __builtin_amdgcn_wmma_f32_16x16x32_f16(false, af, false, bf0,
                                                        (short)0, acc[0], false, false);
        acc[1] = __builtin_amdgcn_wmma_f32_16x16x32_f16(false, af, false, bf1,
                                                        (short)0, acc[1], false, false);
        acc[2] = __builtin_amdgcn_wmma_f32_16x16x32_f16(false, af, false, bf2,
                                                        (short)0, acc[2], false, false);
        acc[3] = __builtin_amdgcn_wmma_f32_16x16x32_f16(false, af, false, bf3,
                                                        (short)0, acc[3], false, false);
        __syncthreads();
    }

    // ---- store: C/D layout col = lane%16, row = v + 8*(lane>=16) ----
    const int rbase = m0 + (wv << 4) + ((lane >> 4) << 3);
#pragma unroll
    for (int j = 0; j < 4; ++j) {
        const int col = n0 + (j << 4) + (lane & 15);
        float bv = (bias != nullptr && col < Nout) ? bias[col] : 0.f;
        if (fullM && fullN) {
#pragma unroll
            for (int v = 0; v < 8; ++v) {
                float o = acc[j][v] + bv;
                if (act == 1) o = fmaxf(o, 0.f);
                C[(size_t)(rbase + v) * ldc + col] = o;
            }
        } else {
#pragma unroll
            for (int v = 0; v < 8; ++v) {
                int r = rbase + v;
                if (r < Mr && col < Nout) {
                    float o = acc[j][v] + bv;
                    if (act == 1) o = fmaxf(o, 0.f);
                    C[(size_t)r * ldc + col] = o;
                }
            }
        }
    }
}

// ======================= elementwise / graph kernels =======================

__global__ void k_fill(float* __restrict__ p, long n, float v) {
    long i = (long)blockIdx.x * blockDim.x + threadIdx.x;
    if (i < n) p[i] = v;
}

__global__ void k_deg(float* __restrict__ deg, const int* __restrict__ col,
                      const float* __restrict__ w, int E_) {
    int e = blockIdx.x * blockDim.x + threadIdx.x;
    if (e < E_) atomicAdd(deg + col[e], w[e]);
}

__global__ void k_rsq(float* __restrict__ p, int n) {
    int i = blockIdx.x * blockDim.x + threadIdx.x;
    if (i >= n) return;
    float d = p[i];
    p[i] = (d > 0.f) ? rsqrtf(fmaxf(d, 1e-12f)) : 0.f;
}

// out[i*ldo+f] = bias[f] + dinv[i]^2 * h[i*F+f]   (self loop + bias init)
__global__ void k_selfloop(float* __restrict__ out, const float* __restrict__ h,
                           const float* __restrict__ dinv,
                           const float* __restrict__ bias,
                           int n, int F, int ldo) {
    long idx = (long)blockIdx.x * blockDim.x + threadIdx.x;
    if (idx >= (long)n * F) return;
    int i = (int)(idx / F), f = (int)(idx % F);
    float d = dinv[i];
    out[(size_t)i * ldo + f] = bias[f] + d * d * h[(size_t)i * F + f];
}

// out[col[e]*ldo + f] += dinv[row]*w*dinv[col] * h[row*F + f], float4 per thread
__global__ void k_scatter(float* __restrict__ out, const float* __restrict__ h,
                          const float* __restrict__ dinv,
                          const int* __restrict__ row, const int* __restrict__ col,
                          const float* __restrict__ w, int E_, int F, int ldo) {
    int fq = F >> 2;
    long total = (long)E_ * fq;
    long idx = (long)blockIdx.x * blockDim.x + threadIdx.x;
    if (idx >= total) return;
    int e  = (int)(idx / fq);
    int f0 = ((int)(idx % fq)) << 2;
    int r = row[e], c = col[e];
    float nm = dinv[r] * w[e] * dinv[c];
    const float4 hv = *reinterpret_cast<const float4*>(h + (size_t)r * F + f0);
    float* o = out + (size_t)c * ldo + f0;
    atomicAdd(o + 0, nm * hv.x);
    atomicAdd(o + 1, nm * hv.y);
    atomicAdd(o + 2, nm * hv.z);
    atomicAdd(o + 3, nm * hv.w);
}

// cat[i][j] += [g2[i](4) | lin2[i](4)] @ P(8x8)
__global__ void k_cat(float* __restrict__ cat, const float* __restrict__ g2,
                      const float* __restrict__ lin2, const float* __restrict__ P,
                      int n) {
    long idx = (long)blockIdx.x * blockDim.x + threadIdx.x;
    if (idx >= (long)n * 8) return;
    int i = (int)(idx >> 3), j = (int)(idx & 7);
    float s = 0.f;
#pragma unroll
    for (int k = 0; k < 4; ++k) s += g2[(size_t)i * 4 + k] * P[k * 8 + j];
#pragma unroll
    for (int k = 0; k < 4; ++k) s += lin2[(size_t)i * 4 + k] * P[(4 + k) * 8 + j];
    cat[idx] += s;
}

// xp[i][g] = bih[g] + xs[i] . Wih[g]   (G=48 gates)
__global__ void k_xp(float* __restrict__ xp, const float* __restrict__ xs,
                     const float* __restrict__ Wih, const float* __restrict__ bih,
                     int n, int K) {
    long idx = (long)blockIdx.x * blockDim.x + threadIdx.x;
    if (idx >= (long)n * 48) return;
    int i = (int)(idx / 48), g = (int)(idx % 48);
    float s = bih[g];
    const float* xr = xs + (size_t)i * K;
    const float* wr = Wih + (size_t)g * K;
    for (int k = 0; k < K; ++k) s += xr[k] * wr[k];
    xp[idx] = s;
}

// Sequential GRU scan: one block, 512 threads = (b,j) over (32,16).
__global__ __launch_bounds__(512) void k_gru_scan(
    const float* __restrict__ xp, const float* __restrict__ Whh,
    const float* __restrict__ bhh, const float* __restrict__ h0,
    float* __restrict__ y, float* __restrict__ hT) {
    __shared__ float hs[32][16];
    __shared__ float Wl[48][16];
    __shared__ float bl[48];
    const int tid = threadIdx.x;
    const int b = tid >> 4, j = tid & 15;
    for (int i2 = tid; i2 < 48 * 16; i2 += 512) Wl[i2 >> 4][i2 & 15] = Whh[i2];
    if (tid < 48) bl[tid] = bhh[tid];
    hs[b][j] = h0[tid];
    __syncthreads();
    for (int t = 0; t < 1000; ++t) {
        const long i = (long)b * 1000 + t;
        float hr = 0.f, hz = 0.f, hn = 0.f;
#pragma unroll
        for (int k = 0; k < 16; ++k) {
            float hk = hs[b][k];
            hr += hk * Wl[j][k];
            hz += hk * Wl[16 + j][k];
            hn += hk * Wl[32 + j][k];
        }
        const float* xpi = xp + (size_t)i * 48;
        float r  = sigmoidf_(xpi[j] + hr + bl[j]);
        float z  = sigmoidf_(xpi[16 + j] + hz + bl[16 + j]);
        float nn = tanhf(xpi[32 + j] + r * (hn + bl[32 + j]));
        float hnew = (1.f - z) * nn + z * hs[b][j];
        __syncthreads();
        hs[b][j] = hnew;
        y[(size_t)i * 16 + j] = hnew;
        __syncthreads();
    }
    hT[tid] = hs[b][j];
}

// o[i] = b[0] + A[i,:K] . w    (K multiple of 4)
__global__ void k_rowdot(float* __restrict__ o, const float* __restrict__ A,
                         const float* __restrict__ w, const float* __restrict__ b,
                         int rows, int K) {
    int i = blockIdx.x * blockDim.x + threadIdx.x;
    if (i >= rows) return;
    float s = b[0];
    const float4* a4 = reinterpret_cast<const float4*>(A + (size_t)i * K);
    const float4* w4 = reinterpret_cast<const float4*>(w);
    for (int k = 0; k < (K >> 2); ++k) {
        float4 av = a4[k], wv = w4[k];
        s += av.x * wv.x + av.y * wv.y + av.z * wv.z + av.w * wv.w;
    }
    o[i] = s;
}

// ro2[(32,2)] = reshape(ro1,(32,4000)) @ W(2x4000)^T + b
__global__ void k_ro2(float* __restrict__ o, const float* __restrict__ r,
                      const float* __restrict__ W, const float* __restrict__ b) {
    int t = threadIdx.x;
    if (t >= 64) return;
    int bt = t >> 1, oo = t & 1;
    float s = b[oo];
    const float* rr = r + (size_t)bt * 4000;
    const float* w  = W + (size_t)oo * 4000;
    for (int j2 = 0; j2 < 4000; ++j2) s += rr[j2] * w[j2];
    o[bt * 2 + oo] = s;
}

__global__ void k_comb(float* __restrict__ o, const float* __restrict__ cfres,
                       const float* __restrict__ ro2, const float* __restrict__ comb) {
    int t = threadIdx.x;
    if (t >= 64) return;
    o[t] = sigmoidf_(cfres[t] * comb[0] + ro2[t] * comb[1]);
}

// dst[i*ldd + f] = src[i*F + f]   (copy rnf into rec_cat columns 64..127)
__global__ void k_copy(float* __restrict__ dst, const float* __restrict__ src,
                       int n, int F, int ldd) {
    long idx = (long)blockIdx.x * blockDim.x + threadIdx.x;
    if (idx >= (long)n * F) return;
    int i = (int)(idx / F), f = (int)(idx % F);
    dst[(size_t)i * ldd + f] = src[idx];
}

// ======================= host side =========================================

static inline unsigned cdiv(long a, int b) { return (unsigned)((a + b - 1) / b); }

static void gemm(const float* A, int lda, const float* W, const float* bias,
                 float* C, int ldc, int Mr, int K, int Nout, int act,
                 hipStream_t s) {
    dim3 g((unsigned)((Mr + BM - 1) / BM), (unsigned)((Nout + BN - 1) / BN));
    k_gemm<<<g, dim3(128), 0, s>>>(A, lda, W, bias, C, ldc, Mr, K, Nout, act);
}

extern "C" void kernel_launch(void* const* d_in, const int* in_sizes, int n_in,
                              void* d_out, int out_size, void* d_ws, size_t ws_size,
                              hipStream_t stream) {
    (void)in_sizes; (void)n_in; (void)out_size; (void)ws_size;
    constexpr int N = 32000, E = 512000;

    const float* x    = (const float*)d_in[0];
    const int*   adj  = (const int*)  d_in[1];
    const float* wei  = (const float*)d_in[2];
    const float* Hin  = (const float*)d_in[3];
    // params dict flattened with JAX pytree (sorted key) order:
    const float* Wg1  = (const float*)d_in[4];
    const float* Wg2  = (const float*)d_in[5];
    const float* Whh0 = (const float*)d_in[6];
    const float* Whh1 = (const float*)d_in[7];
    const float* Wih0 = (const float*)d_in[8];
    const float* Wih1 = (const float*)d_in[9];
    const float* Wr1  = (const float*)d_in[10];
    const float* Wr2  = (const float*)d_in[11];
    const float* bg1  = (const float*)d_in[12];
    const float* bg2  = (const float*)d_in[13];
    const float* bhh0 = (const float*)d_in[14];
    const float* bhh1 = (const float*)d_in[15];
    const float* bih0 = (const float*)d_in[16];
    const float* bih1 = (const float*)d_in[17];
    const float* br1  = (const float*)d_in[18];
    const float* br2  = (const float*)d_in[19];
    const float* catP = (const float*)d_in[20];
    const float* cfW1 = (const float*)d_in[21];
    const float* cfW2 = (const float*)d_in[22];
    const float* cfW3 = (const float*)d_in[23];
    const float* cfb1 = (const float*)d_in[24];
    const float* cfb2 = (const float*)d_in[25];
    const float* cfb3 = (const float*)d_in[26];
    const float* comb = (const float*)d_in[27];
    const float* mlW1 = (const float*)d_in[28];
    const float* mlW2 = (const float*)d_in[29];
    const float* mlb1 = (const float*)d_in[30];
    const float* mlb2 = (const float*)d_in[31];
    const float* nfW1 = (const float*)d_in[32];
    const float* nfW2 = (const float*)d_in[33];
    const float* nfb1 = (const float*)d_in[34];
    const float* nfb2 = (const float*)d_in[35];
    const float* recW = (const float*)d_in[36];
    const float* recb = (const float*)d_in[37];
    const float* rnfW1 = (const float*)d_in[38];
    const float* rnfW2 = (const float*)d_in[39];
    const float* rnfb1 = (const float*)d_in[40];
    const float* rnfb2 = (const float*)d_in[41];
    const float* roW1 = (const float*)d_in[42];
    const float* roW2 = (const float*)d_in[43];
    const float* rob1 = (const float*)d_in[44];
    const float* rob2 = (const float*)d_in[45];

    float* out = (float*)d_out;
    float* ws  = (float*)d_ws;

    // -------- workspace layout (floats) --------
    float* dinvA  = ws;                            // 4*N
    float* h192   = dinvA  + (size_t)4 * N;        // N*192
    float* g192   = h192   + (size_t)N * 192;      // N*192
    float* b128   = g192   + (size_t)N * 192;      // N*128
    float* sm     = b128   + (size_t)N * 128;      // N*16
    float* h2     = sm;
    float* g2     = sm + (size_t)N * 4;
    float* lin2   = sm + (size_t)N * 8;
    float* cat    = sm + (size_t)N * 16;           // N*8
    float* reline = cat    + (size_t)N * 8;        // N*64
    float* rnf    = reline + (size_t)N * 64;       // N*64
    float* reccat = rnf    + (size_t)N * 64;       // 4N*128
    float* xp     = reccat + (size_t)4 * N * 128;  // N*48
    float* y0     = xp     + (size_t)N * 48;       // N*16
    float* y1     = y0     + (size_t)N * 16;       // N*16
    float* cf1    = y1     + (size_t)N * 16;       // 32*256
    float* cf2    = cf1 + 32 * 256;                // 32*64
    float* cfres  = cf2 + 32 * 64;                 // 32*2
    float* ro1    = cfres + 64;                    // 4N
    float* ro2    = ro1 + (size_t)4 * N;           // 64

    float* rec_out = out + 64;                       // 4N*64
    float* newH    = out + 64 + (size_t)4 * N * 64;  // 2*32*16

    const int TB = 256;

    // -------- phase 1: encoder GCN + node MLP, accumulate cat --------
    k_fill<<<cdiv((long)N * 8, TB), TB, 0, stream>>>(cat, (long)N * 8, 0.f);

    for (int m = 0; m < 4; ++m) {
        const int*   rowm = adj + (size_t)m * 2 * E;
        const int*   colm = rowm + E;
        const float* wm   = wei + (size_t)m * E;
        float*       dinv = dinvA + (size_t)m * N;

        k_fill<<<cdiv(N, TB), TB, 0, stream>>>(dinv, N, 1.f);      // self-loops
        k_deg <<<cdiv(E, TB), TB, 0, stream>>>(dinv, colm, wm, E);
        k_rsq <<<cdiv(N, TB), TB, 0, stream>>>(dinv, N);

        const float* xm = x + (size_t)m * N * 64;
        // conv1: h = x @ Wg1^T ; g = A_norm h + bg1
        gemm(xm, 64, Wg1 + (size_t)m * 192 * 64, nullptr, h192, 192, N, 64, 192, 0, stream);
        k_selfloop<<<cdiv((long)N * 192, TB), TB, 0, stream>>>(g192, h192, dinv, bg1 + m * 192, N, 192, 192);
        k_scatter <<<cdiv((long)E * 48, TB), TB, 0, stream>>>(g192, h192, dinv, rowm, colm, wm, E, 192, 192);
        // conv2
        gemm(g192, 192, Wg2 + (size_t)m * 4 * 192, nullptr, h2, 4, N, 192, 4, 0, stream);
        k_selfloop<<<cdiv((long)N * 4, TB), TB, 0, stream>>>(g2, h2, dinv, bg2 + m * 4, N, 4, 4);
        k_scatter <<<cdiv((long)E * 1, TB), TB, 0, stream>>>(g2, h2, dinv, rowm, colm, wm, E, 4, 4);
        // node MLP
        gemm(xm, 64, nfW1, nfb1, b128, 128, N, 64, 128, 1, stream);
        gemm(b128, 128, nfW2, nfb2, lin2, 4, N, 128, 4, 0, stream);
        // cat += [g|lin] @ catP[m]
        k_cat<<<cdiv((long)N * 8, TB), TB, 0, stream>>>(cat, g2, lin2, catP + (size_t)m * 64, N);
    }

    // -------- phase 2: 2-layer GRU --------
    k_xp<<<cdiv((long)N * 48, TB), TB, 0, stream>>>(xp, cat, Wih0, bih0, N, 8);
    k_gru_scan<<<1, 512, 0, stream>>>(xp, Whh0, bhh0, Hin, y0, newH);
    k_xp<<<cdiv((long)N * 48, TB), TB, 0, stream>>>(xp, y0, Wih1, bih1, N, 16);
    k_gru_scan<<<1, 512, 0, stream>>>(xp, Whh1, bhh1, Hin + 512, y1, newH + 512);

    // -------- phase 3: classifier head (flat = y1 viewed as 32 x 16000) ----
    gemm(y1, 16000, cfW1, cfb1, cf1, 256, 32, 16000, 256, 1, stream);
    gemm(cf1, 256, cfW2, cfb2, cf2, 64, 32, 256, 64, 1, stream);
    gemm(cf2, 64, cfW3, cfb3, cfres, 2, 32, 64, 2, 0, stream);

    // -------- phase 4: re_line and rnf --------
    gemm(cat, 8, mlW1, mlb1, b128, 128, N, 8, 128, 1, stream);
    gemm(b128, 128, mlW2, mlb2, reline, 64, N, 128, 64, 0, stream);
    gemm(reline, 64, rnfW1, rnfb1, b128, 128, N, 64, 128, 1, stream);
    gemm(b128, 128, rnfW2, rnfb2, rnf, 64, N, 128, 64, 0, stream);

    // -------- phase 5: reconstruction GCN stacks, fused concat ------------
    for (int m = 0; m < 4; ++m) {
        const int*   rowm = adj + (size_t)m * 2 * E;
        const int*   colm = rowm + E;
        const float* wm   = wei + (size_t)m * E;
        float*       dinv = dinvA + (size_t)m * N;

        gemm(reline, 64, Wr1 + (size_t)m * 192 * 64, nullptr, h192, 192, N, 64, 192, 0, stream);
        k_selfloop<<<cdiv((long)N * 192, TB), TB, 0, stream>>>(g192, h192, dinv, br1 + m * 192, N, 192, 192);
        k_scatter <<<cdiv((long)E * 48, TB), TB, 0, stream>>>(g192, h192, dinv, rowm, colm, wm, E, 192, 192);

        gemm(g192, 192, Wr2 + (size_t)m * 64 * 192, nullptr, b128, 64, N, 192, 64, 0, stream);
        float* rc = reccat + (size_t)m * N * 128;
        k_selfloop<<<cdiv((long)N * 64, TB), TB, 0, stream>>>(rc, b128, dinv, br2 + m * 64, N, 64, 128);
        k_scatter <<<cdiv((long)E * 16, TB), TB, 0, stream>>>(rc, b128, dinv, rowm, colm, wm, E, 64, 128);
        k_copy    <<<cdiv((long)N * 64, TB), TB, 0, stream>>>(rc + 64, rnf, N, 64, 128);
    }

    // -------- phase 6: rec_out + readout + combine --------
    gemm(reccat, 128, recW, recb, rec_out, 64, 4 * N, 128, 64, 0, stream);
    k_rowdot<<<cdiv((long)4 * N, TB), TB, 0, stream>>>(ro1, rec_out, roW1, rob1, 4 * N, 64);
    k_ro2 <<<1, 64, 0, stream>>>(ro2, ro1, roW2, rob2);
    k_comb<<<1, 64, 0, stream>>>(out, cfres, ro2, comb);
}
